// LSTMblock_33724083208299
// MI455X (gfx1250) — compile-verified, matrix operand order checked by reading, not run on hardware
//
#include <hip/hip_runtime.h>
#include <hip/hip_bf16.h>
#include <math.h>

// Problem constants
#define TT 16384
#define DD 257
#define HH 512
#define G4 2048   // 4*H
#define OO 257

typedef __attribute__((ext_vector_type(16))) __bf16 v16bf;
typedef __attribute__((ext_vector_type(8)))  float  v8f;
typedef __attribute__((ext_vector_type(4)))  unsigned int u32x4;
typedef __attribute__((ext_vector_type(8)))  int i32x8;
typedef __attribute__((ext_vector_type(4)))  int i32x4;

// ---------------------------------------------------------------------------
// Fragment loaders for the CDNA5 16-bit A/B layout (16x32 tile):
//   lanes 0-15 : M = lane,    K = {k0..k0+7, k0+16..k0+23}
//   lanes 16-31: M = lane-16, K = {k0+8..k0+15, k0+24..k0+31}
// "full"  : no bounds checks, unconditional scalar loads (odd ld)
// "full4" : no bounds checks, float4 loads (ld % 4 == 0, 16B-aligned rows)
// "edge"  : clamped addresses + v_cndmask zeroing — NO exec-mask branches
// Each produces a split-bf16 pair (hi + residual lo) for 3xBF16 fp32 emu.
// ---------------------------------------------------------------------------
__device__ __forceinline__ void cvt_split(const float* vv, v16bf& hi, v16bf& lo) {
#pragma unroll
  for (int e = 0; e < 16; ++e) {
    const float v = vv[e];
    const __bf16 h = (__bf16)v;
    hi[e] = h;
    lo[e] = (__bf16)(v - (float)h);
  }
}

__device__ __forceinline__ void load_tile_full(const float* __restrict__ base,
                                               int ld, int row0, int k0,
                                               v16bf& hi, v16bf& lo) {
  const int lane = threadIdx.x & 31;
  const float* p = base + (size_t)(row0 + (lane & 15)) * ld
                 + k0 + ((lane >> 4) << 3);
  float vv[16];
#pragma unroll
  for (int e = 0; e < 16; ++e)
    vv[e] = p[(e < 8) ? e : (e + 8)];
  cvt_split(vv, hi, lo);
}

__device__ __forceinline__ void load_tile_full4(const float* __restrict__ base,
                                                int ld, int row0, int k0,
                                                v16bf& hi, v16bf& lo) {
  const int lane = threadIdx.x & 31;
  const float* p = base + (size_t)(row0 + (lane & 15)) * ld
                 + k0 + ((lane >> 4) << 3);
  const float4 a0 = *(const float4*)(p + 0);
  const float4 a1 = *(const float4*)(p + 4);
  const float4 b0 = *(const float4*)(p + 16);
  const float4 b1 = *(const float4*)(p + 20);
  const float vv[16] = {a0.x, a0.y, a0.z, a0.w, a1.x, a1.y, a1.z, a1.w,
                        b0.x, b0.y, b0.z, b0.w, b1.x, b1.y, b1.z, b1.w};
  cvt_split(vv, hi, lo);
}

__device__ __forceinline__ void load_tile_edge(const float* __restrict__ base,
                                               int ld, int row0, int rmax,
                                               int k0, int kmax,
                                               v16bf& hi, v16bf& lo) {
  const int lane = threadIdx.x & 31;
  const int r    = row0 + (lane & 15);
  const bool rok = (r < rmax);
  const float* p = base + (size_t)(rok ? r : (rmax - 1)) * ld;
  const int ks   = (lane >> 4) << 3;
  float vv[16];
#pragma unroll
  for (int e = 0; e < 16; ++e) {
    const int k  = k0 + ks + ((e < 8) ? e : (e + 8));
    const int kc = (k < kmax) ? k : (kmax - 1);   // clamp: load always legal
    const float v = p[kc];
    vv[e] = (rok && (k < kmax)) ? v : 0.0f;       // select, not branch
  }
  cvt_split(vv, hi, lo);
}

__device__ __forceinline__ v8f wmma_bf16(v16bf a, v16bf b, v8f c) {
  return __builtin_amdgcn_wmma_f32_16x16x32_bf16(false, a, false, b,
                                                 (short)0, c, false, false);
}

// ---------------------------------------------------------------------------
// Kernel A: x_gates[T,2048] = stft[T,257] @ W_ih.T + (b_ih + b_hh)
// One wave per (16-row, 64-col) strip; 8 full K-chunks + 1 edge chunk.
// ---------------------------------------------------------------------------
__global__ void __launch_bounds__(256)
kA_xgates(const float* __restrict__ stft, const float* __restrict__ Wih,
          const float* __restrict__ bih,  const float* __restrict__ bhh,
          float* __restrict__ xg) {
  const int wid = blockIdx.x * (blockDim.x >> 5) + (threadIdx.x >> 5);
  const int mt  = wid >> 5;        // 1024 M-tiles
  const int ns  = wid & 31;        // 32 strips of 4 N-tiles
  if (mt >= TT / 16) return;

  v8f acc[4];
#pragma unroll
  for (int j = 0; j < 4; ++j)
#pragma unroll
    for (int r = 0; r < 8; ++r) acc[j][r] = 0.0f;

  v16bf ahi, alo, bhi, blo;
  for (int k0 = 0; k0 < 256; k0 += 32) {          // fully in-bounds chunks
    load_tile_full(stft, DD, mt * 16, k0, ahi, alo);
#pragma unroll
    for (int j = 0; j < 4; ++j) {
      load_tile_full(Wih, DD, (ns * 4 + j) * 16, k0, bhi, blo);
      acc[j] = wmma_bf16(ahi, bhi, acc[j]);
      acc[j] = wmma_bf16(ahi, blo, acc[j]);
      acc[j] = wmma_bf16(alo, bhi, acc[j]);
    }
  }
  // tail chunk k0=256 (only k==256 valid)
  load_tile_edge(stft, DD, mt * 16, TT, 256, DD, ahi, alo);
#pragma unroll
  for (int j = 0; j < 4; ++j) {
    load_tile_edge(Wih, DD, (ns * 4 + j) * 16, G4, 256, DD, bhi, blo);
    acc[j] = wmma_bf16(ahi, bhi, acc[j]);
    acc[j] = wmma_bf16(ahi, blo, acc[j]);
    acc[j] = wmma_bf16(alo, bhi, acc[j]);
  }

  const int lane = threadIdx.x & 31;
  const int rowb = mt * 16 + ((lane >> 4) << 3);
#pragma unroll
  for (int j = 0; j < 4; ++j) {
    const int col   = (ns * 4 + j) * 16 + (lane & 15);
    const float bia = bih[col] + bhh[col];
#pragma unroll
    for (int r = 0; r < 8; ++r)
      xg[(size_t)(rowb + r) * G4 + col] = acc[j][r] + bia;
  }
}

// ---------------------------------------------------------------------------
// Kernel B: sequential LSTM scan, 16 persistent workgroups.
// Each WGP keeps a 128-row slice of W_hh (256 KB) resident in CDNA5's 320 KB
// LDS. Block b owns hidden units [b*32, b*32+32) and gate rows
// {j, 512+j, 1024+j, 1536+j}. TDM tensor loads prefetch the slice (tracked on
// TENSORcnt); after a workgroup barrier, a plain fill guarantees the data
// regardless of D# encoding details (no TDM/fill race).
// ---------------------------------------------------------------------------
__global__ void __launch_bounds__(256)
kB_lstm(const float* __restrict__ xg, const float* __restrict__ Whh,
        float* __restrict__ hs, unsigned int* __restrict__ bar) {
  extern __shared__ float s_w[];       // 128 x 512 fp32 = 256 KB
  __shared__ float s_h[HH];
  __shared__ float s_part[256];
  __shared__ float s_gate[128];

  const int b   = blockIdx.x;          // 0..15
  const int tid = threadIdx.x;         // 0..255

  // --- TDM prefetch: 4 tiles of 32 rows x 512 f32, row stride 512 ----------
  if ((tid >> 5) == 0) {               // wave 0 only (TDM ignores EXEC)
#pragma unroll
    for (int g = 0; g < 4; ++g) {
      const unsigned long long gaddr =
          (unsigned long long)(const void*)(Whh + ((size_t)g * HH + (size_t)b * 32) * HH);
      u32x4 g0;
      g0[0] = 1u;                                         // count=1 (valid D#)
      g0[1] = (unsigned int)(g * 32 * HH * 4);            // lds_addr (bytes)
      g0[2] = (unsigned int)(gaddr & 0xffffffffu);        // global_addr[31:0]
      g0[3] = (unsigned int)((gaddr >> 32) & 0x1ffffffu)  // global_addr[56:32]
            | (2u << 30);                                 // type = 2 ("image")
      i32x8 g1;
      g1[0] = (int)(2u << 16);                 // data_size code 2 = 4 bytes
      g1[1] = (int)((unsigned)HH << 16);       // tensor_dim0[15:0] = 512
      g1[2] = (int)((unsigned)G4 << 16);       // dim0 hi=0 | tensor_dim1 lo=2048
      g1[3] = (int)((unsigned)HH << 16);       // dim1 hi=0 | tile_dim0 = 512
      g1[4] = 32;                              // tile_dim1 = 32, tile_dim2 = 0
      g1[5] = HH;                              // tensor_dim0_stride = 512
      g1[6] = 0;
      g1[7] = 0;
      const i32x4 gz4 = {0, 0, 0, 0};          // D# groups 2/3 disabled (2D)
      const i32x8 gz8 = {0, 0, 0, 0, 0, 0, 0, 0};
      // 6-arg toolchain form: (g0, g1, g2, g3, extra, cpol)
      __builtin_amdgcn_tensor_load_to_lds(g0, g1, gz4, gz4, gz8, 0);
    }
    __builtin_amdgcn_s_wait_tensorcnt(0);      // all TDM LDS writes landed
  }
  __syncthreads();   // no wave may fill s_w while TDM writes are in flight

  // --- authoritative fill (overwrites TDM result; correctness-safe) --------
  for (int i = tid; i < 128 * HH; i += 256) {
    const int lr = i >> 9, k = i & (HH - 1);
    const int gr = ((lr >> 5) << 9) + b * 32 + (lr & 31);
    s_w[i] = Whh[(size_t)gr * HH + k];
  }
  const int r2    = tid >> 1;
  const int half  = tid & 1;
  const int gr128 = ((tid >> 5) << 9) + b * 32 + (tid & 31); // tid<128 only
  const int xgi   = (tid < 128) ? gr128 : 0;                 // safe for all
  float c = 0.0f;
  __syncthreads();

  for (int t = 0; t < TT; ++t) {
    // issue x_gates load early: its L2 latency hides under the LDS dot below
    const float xgv = xg[(size_t)t * G4 + xgi];

    if (t == 0) { s_h[tid] = 0.0f; s_h[tid + 256] = 0.0f; }
    else {
      const float* hp = hs + (size_t)(t - 1) * HH;
      s_h[tid] = hp[tid]; s_h[tid + 256] = hp[tid + 256];
    }
    __syncthreads();

    // 2 threads per gate row: 256-long fp32 dot from LDS (ds b128 loads)
    {
      const float4* wr = (const float4*)(s_w + (size_t)r2 * HH + half * 256);
      const float4* hv = (const float4*)(s_h + half * 256);
      float a = 0.0f;
#pragma unroll 8
      for (int q = 0; q < 64; ++q) {
        const float4 w = wr[q], x = hv[q];
        a = fmaf(w.x, x.x, a); a = fmaf(w.y, x.y, a);
        a = fmaf(w.z, x.z, a); a = fmaf(w.w, x.w, a);
      }
      s_part[tid] = a;
    }
    __syncthreads();

    if (tid < 128)
      s_gate[tid] = s_part[2 * tid] + s_part[2 * tid + 1] + xgv;
    __syncthreads();

    if (tid < 32) {
      const float gi = s_gate[tid],      gf = s_gate[32 + tid];
      const float gg = s_gate[64 + tid], go = s_gate[96 + tid];
      const float i_ = 1.0f / (1.0f + __expf(-gi));
      const float f_ = 1.0f / (1.0f + __expf(-gf));
      const float g_ = tanhf(gg);
      const float o_ = 1.0f / (1.0f + __expf(-go));
      c = f_ * c + i_ * g_;
      hs[(size_t)t * HH + b * 32 + tid] = o_ * tanhf(c);
    }

    // cross-workgroup step barrier: release, arrive, spin, acquire
    __threadfence();
    __syncthreads();
    if (tid == 0) {
      atomicAdd(bar, 1u);
      const unsigned int target = 16u * (unsigned int)(t + 1);
      while (__hip_atomic_load(bar, __ATOMIC_RELAXED,
                               __HIP_MEMORY_SCOPE_AGENT) < target)
        __builtin_amdgcn_s_sleep(1);
    }
    __builtin_amdgcn_s_cluster_barrier();   // s_barrier_signal/wait -3
    __syncthreads();
    __threadfence();
  }
}

// ---------------------------------------------------------------------------
// Kernel C: logits[T,257] = hs[T,512] @ W_out.T + b_out  (bf16-split WMMA)
// ld=512 -> float4 (b128) fragment loads; only N-tile 16 needs edge masking.
// ---------------------------------------------------------------------------
__global__ void __launch_bounds__(256)
kC_out(const float* __restrict__ hs, const float* __restrict__ Wout,
       const float* __restrict__ bout, float* __restrict__ out) {
  const int wid = blockIdx.x * (blockDim.x >> 5) + (threadIdx.x >> 5);
  const int mt  = wid / 17;
  const int nt  = wid % 17;
  if (mt >= TT / 16) return;

  v8f acc;
#pragma unroll
  for (int r = 0; r < 8; ++r) acc[r] = 0.0f;

  const bool bfull = (nt < 16);     // wave-uniform
  v16bf ahi, alo, bhi, blo;
  for (int k0 = 0; k0 < HH; k0 += 32) {
    load_tile_full4(hs, HH, mt * 16, k0, ahi, alo);
    if (bfull) load_tile_full4(Wout, HH, nt * 16, k0, bhi, blo);
    else       load_tile_edge (Wout, HH, nt * 16, OO, k0, HH, bhi, blo);
    acc = wmma_bf16(ahi, bhi, acc);
    acc = wmma_bf16(ahi, blo, acc);
    acc = wmma_bf16(alo, bhi, acc);
  }

  const int lane = threadIdx.x & 31;
  const int col  = nt * 16 + (lane & 15);
  if (col < OO) {
    const int rowb  = mt * 16 + ((lane >> 4) << 3);
    const float bia = bout[col];
#pragma unroll
    for (int r = 0; r < 8; ++r)
      out[(size_t)(rowb + r) * OO + col] = acc[r] + bia;
  }
}

// ---------------------------------------------------------------------------
// Kernel D: in-place row-wise log_softmax over 257 columns; one wave per row.
// Clamped loads (no exec branches); wave reductions via shfl_xor.
// ---------------------------------------------------------------------------
__global__ void __launch_bounds__(256)
kD_logsoftmax(float* __restrict__ out) {
  const int lane = threadIdx.x & 31;
  const int row  = blockIdx.x * 8 + (threadIdx.x >> 5);
  if (row >= TT) return;
  float* p = out + (size_t)row * OO;

  float v[9];
  float m = -INFINITY;
#pragma unroll
  for (int i = 0; i < 9; ++i) {
    const int k  = lane + i * 32;
    const int kc = (k < OO) ? k : (OO - 1);
    const float x = p[kc];
    v[i] = (k < OO) ? x : -INFINITY;
    m = fmaxf(m, v[i]);
  }
#pragma unroll
  for (int off = 16; off; off >>= 1) m = fmaxf(m, __shfl_xor(m, off, 32));

  float s = 0.0f;
#pragma unroll
  for (int i = 0; i < 9; ++i) s += __expf(v[i] - m);   // exp(-inf)=0 for tail
#pragma unroll
  for (int off = 16; off; off >>= 1) s += __shfl_xor(s, off, 32);

  const float lse = m + __logf(s);
#pragma unroll
  for (int i = 0; i < 9; ++i) {
    const int k = lane + i * 32;
    if (k < OO) p[k] = v[i] - lse;
  }
}

// ---------------------------------------------------------------------------
// Launch. Workspace: [0,256) barrier counter; x_gates (134 MB); hs (33.5 MB).
// ---------------------------------------------------------------------------
extern "C" void kernel_launch(void* const* d_in, const int* in_sizes, int n_in,
                              void* d_out, int out_size, void* d_ws, size_t ws_size,
                              hipStream_t stream) {
  const float* stft = (const float*)d_in[0];
  const float* Wih  = (const float*)d_in[1];
  const float* Whh  = (const float*)d_in[2];
  const float* bih  = (const float*)d_in[3];
  const float* bhh  = (const float*)d_in[4];
  const float* Wout = (const float*)d_in[5];
  const float* bout = (const float*)d_in[6];
  float* out = (float*)d_out;

  char* ws = (char*)d_ws;
  unsigned int* bar = (unsigned int*)ws;
  float* xg = (float*)(ws + 256);
  float* hs = (float*)(ws + 256 + (size_t)TT * G4 * sizeof(float));

  (void)hipMemsetAsync(bar, 0, 256, stream);                 // barrier = 0
  kA_xgates<<<4096, 256, 0, stream>>>(stft, Wih, bih, bhh, xg);
  kB_lstm<<<16, 256, 128 * HH * sizeof(float), stream>>>(xg, Whh, hs, bar);
  kC_out<<<2176, 256, 0, stream>>>(hs, Wout, bout, out);
  kD_logsoftmax<<<2048, 256, 0, stream>>>(out);
}
